// TreeRecurrentTagger_71717363909240
// MI455X (gfx1250) — compile-verified, matrix operand order checked by reading, not run on hardware
//
#include <hip/hip_runtime.h>
#include <hip/hip_bf16.h>

// Problem constants (from reference)
#define B_    32
#define L_    512
#define H_    1024
#define SH_   50
#define C_    3
#define N_    1023          // 2*L-1
#define LVLS_ 9             // log2(L)

// Padded K dims (multiple of 32 for WMMA K-chunks)
#define KL_P 1088   // leaf:  H+C+SH = 1077
#define KC_P 2112   // comp:  2H+SH  = 2098
#define KG_P 1088   // grev:  H+SH   = 1074
#define KD_P 2176   // dec:   2H+2SH = 2148

typedef __attribute__((ext_vector_type(16))) __bf16 v16bf;
typedef __attribute__((ext_vector_type(8)))  float  v8f;

static __device__ __forceinline__ unsigned short f2bf(float f) {
  union { float f; unsigned u; } x; x.f = f;
  unsigned r = x.u + 0x7FFFu + ((x.u >> 16) & 1u);   // round-to-nearest-even
  return (unsigned short)(r >> 16);
}
static __device__ __forceinline__ float bf2f(unsigned short h) {
  union { unsigned u; float f; } x; x.u = ((unsigned)h) << 16;
  return x.f;
}

// CDNA5 async global->LDS copy (ASYNCcnt). vDst = per-lane LDS byte address,
// vAddr = per-lane 64-bit global address; 16 bytes per lane.
static __device__ __forceinline__ void async_copy16(unsigned lds_addr,
                                                    const unsigned short* g) {
  asm volatile("global_load_async_to_lds_b128 %0, %1, off"
               :: "v"(lds_addr), "v"(g) : "memory");
}

#if defined(__has_builtin)
#if __has_builtin(__builtin_amdgcn_s_wait_asynccnt)
#define WAIT_ASYNC(n) __builtin_amdgcn_s_wait_asynccnt(n)
#endif
#endif
#ifndef WAIT_ASYNC
#define WAIT_ASYNC(n) asm volatile("s_wait_asynccnt " #n ::: "memory")
#endif

// ---------------------------------------------------------------------------
// Weight transpose + bf16 convert:  W (K x Nout, f32)  ->  Wt (Nout x Kp, bf16)
// ---------------------------------------------------------------------------
__global__ void wT_kernel(const float* __restrict__ W, unsigned short* __restrict__ Wt,
                          int K, int Nout, int Kp) {
  long long i = (long long)blockIdx.x * blockDim.x + threadIdx.x;
  long long total = (long long)Nout * Kp;
  if (i >= total) return;
  int n = (int)(i / Kp), k = (int)(i % Kp);
  Wt[i] = (k < K) ? f2bf(W[(long long)k * Nout + n]) : (unsigned short)0;
}

// ---------------------------------------------------------------------------
// Ctx builders (write bf16 row-major A operands, zero-padded to Kp)
// ---------------------------------------------------------------------------
__global__ void leaf_build(const int* __restrict__ x, const int* __restrict__ wi,
                           const int* __restrict__ cue, const int* __restrict__ pos,
                           const float* __restrict__ wordW, const float* __restrict__ synW,
                           unsigned short* __restrict__ A) {
  int r = blockIdx.x;                 // 0 .. B*L-1
  int b = r / L_, i = r % L_;
  int w = wi[b * L_ + i];
  int s = x[b * N_ + w];
  int c = cue[b * N_ + w];
  int p = pos[b * L_ + i];
  unsigned short* row = A + (long long)r * KL_P;
  for (int j = threadIdx.x; j < KL_P; j += blockDim.x) {
    float v;
    if (j < H_)                 v = wordW[(long long)s * H_ + j];
    else if (j < H_ + C_)       v = (j - H_ == c) ? 1.0f : 0.0f;
    else if (j < H_ + C_ + SH_) v = synW[p * SH_ + (j - H_ - C_)];
    else                        v = 0.0f;
    row[j] = f2bf(v);
  }
}

__global__ void comp_build(const int* __restrict__ x, const float* __restrict__ synW,
                           const unsigned short* __restrict__ up,
                           unsigned short* __restrict__ A, int first, int nn) {
  int r = blockIdx.x; int b = r / nn; int node = first + r % nn;
  int l = 2 * node + 1, rr = 2 * node + 2;
  const unsigned short* upl = up + ((long long)b * N_ + l)  * H_;
  const unsigned short* upr = up + ((long long)b * N_ + rr) * H_;
  int s = x[b * N_ + node];
  unsigned short* row = A + (long long)r * KC_P;
  for (int j = threadIdx.x; j < KC_P; j += blockDim.x) {
    unsigned short v;
    if (j < SH_)               v = f2bf(synW[s * SH_ + j]);
    else if (j < SH_ + H_)     v = upl[j - SH_];
    else if (j < SH_ + 2 * H_) v = upr[j - SH_ - H_];
    else                       v = 0;
    row[j] = v;
  }
}

__global__ void root_build(const int* __restrict__ x, const float* __restrict__ synW,
                           const unsigned short* __restrict__ up,
                           unsigned short* __restrict__ A) {
  int b = blockIdx.x;
  const unsigned short* u0 = up + (long long)b * N_ * H_;
  int s = x[b * N_];
  unsigned short* row = A + (long long)b * KG_P;
  for (int j = threadIdx.x; j < KG_P; j += blockDim.x) {
    unsigned short v;
    if (j < H_)            v = u0[j];
    else if (j < H_ + SH_) v = f2bf(synW[s * SH_ + (j - H_)]);
    else                   v = 0;
    row[j] = v;
  }
}

__global__ void dec_build(const int* __restrict__ x, const int* __restrict__ pos,
                          const float* __restrict__ synW,
                          const unsigned short* __restrict__ up,
                          const unsigned short* __restrict__ down,
                          unsigned short* __restrict__ A,
                          int first, int nn, int isLeafLvl) {
  int r = blockIdx.x; int b = r / nn; int node = first + r % nn;
  int l = 2 * node + 1, rr = 2 * node + 2;
  int sl = isLeafLvl ? pos[b * L_ + (l  - (L_ - 1))] : x[b * N_ + l];
  int sr = isLeafLvl ? pos[b * L_ + (rr - (L_ - 1))] : x[b * N_ + rr];
  const unsigned short* dn = down + ((long long)b * N_ + node) * H_;
  const unsigned short* un = up   + ((long long)b * N_ + node) * H_;
  unsigned short* row = A + (long long)r * KD_P;
  for (int j = threadIdx.x; j < KD_P; j += blockDim.x) {
    unsigned short v;
    if (j < SH_)                   v = f2bf(synW[sl * SH_ + j]);
    else if (j < 2 * SH_)          v = f2bf(synW[sr * SH_ + (j - SH_)]);
    else if (j < 2 * SH_ + H_)     v = dn[j - 2 * SH_];
    else if (j < 2 * SH_ + 2 * H_) v = un[j - 2 * SH_ - H_];
    else                           v = 0;
    row[j] = v;
  }
}

__global__ void dec_scatter(const unsigned short* __restrict__ tmp,
                            unsigned short* __restrict__ down, int first, int nn) {
  int r = blockIdx.x; int b = r / nn; int node = first + r % nn;
  int l = 2 * node + 1, rr = 2 * node + 2;
  const unsigned short* src = tmp + (long long)r * (2 * H_);
  unsigned short* dl = down + ((long long)b * N_ + l)  * H_;
  unsigned short* dr = down + ((long long)b * N_ + rr) * H_;
  for (int j = threadIdx.x; j < H_; j += blockDim.x) {
    dl[j] = src[j];
    dr[j] = src[H_ + j];
  }
}

// ---------------------------------------------------------------------------
// Generic bf16 WMMA GEMM:  C(M x Nout) = A(M x Kp) * Bt(Nout x Kp)^T + bias
//   - 256 threads / 8 waves, 128x128 tile, K-chunk 32
//   - double-buffered LDS tiles filled with global_load_async_to_lds_b128
//     (ASYNCcnt), compute overlaps next-chunk DMA
//   - M-tail rows clamped at the source (A row r only feeds C row r, and
//     OOB C rows are never stored), so EXEC stays uniform for async+WMMA
//   - wave (wm in 0..3, wn in 0..1) computes 32x64 via 2x4 v_wmma 16x16x32
// ---------------------------------------------------------------------------
#define TM 128
#define TN 128
#define LSTR 48   // LDS row stride in halves (96B, 16B-aligned)

__global__ __launch_bounds__(256)
void gemm_bf16_wmma(const unsigned short* __restrict__ A,
                    const unsigned short* __restrict__ Bt,
                    const float* __restrict__ bias,
                    void* __restrict__ Out,
                    int Mrows, int Kp,
                    int act, int outIsBf16,
                    int nn, long long strideB, long long strideRow) {
  __shared__ unsigned short la[2][TM][LSTR];
  __shared__ unsigned short lb[2][TN][LSTR];

  const int tid  = threadIdx.x;
  const int lane = tid & 31;
  const int wid  = tid >> 5;
  const int wm   = wid & 3;      // row-group of 32
  const int wn   = wid >> 2;     // col-group of 64
  const int tileM = blockIdx.y * TM;
  const int tileN = blockIdx.x * TN;

  v8f acc[2][4];
  #pragma unroll
  for (int i = 0; i < 2; i++)
    #pragma unroll
    for (int j = 0; j < 4; j++)
      #pragma unroll
      for (int e = 0; e < 8; e++) acc[i][j][e] = 0.0f;

  const int ldRow = tid >> 1;            // 0..127
  const int ldSeg = (tid & 1) * 16;      // halves 0 or 16

  // clamped A row: garbage rows only pollute discarded C rows
  int gra = tileM + ldRow; if (gra >= Mrows) gra = Mrows - 1;
  const unsigned short* aRow = A  + (long long)gra * Kp + ldSeg;
  const unsigned short* bRow = Bt + (long long)(tileN + ldRow) * Kp + ldSeg;

  const int fl = lane & 15;
  const int kb = (lane < 16) ? 0 : 8;    // ISA 16-bit A/B fragment K-base

  const int nCh = Kp >> 5;               // K chunks of 32

  // Issue chunk `c` into buffer `buf` (4 async b128 per thread, uniform EXEC)
  auto issue = [&](int c, int buf) {
    const unsigned short* ga = aRow + c * 32;
    const unsigned short* gb = bRow + c * 32;
    async_copy16((unsigned)(unsigned long long)&la[buf][ldRow][ldSeg],     ga);
    async_copy16((unsigned)(unsigned long long)&la[buf][ldRow][ldSeg + 8], ga + 8);
    async_copy16((unsigned)(unsigned long long)&lb[buf][ldRow][ldSeg],     gb);
    async_copy16((unsigned)(unsigned long long)&lb[buf][ldRow][ldSeg + 8], gb + 8);
  };

  issue(0, 0);

  for (int i = 0; i < nCh; ++i) {
    const int buf = i & 1;
    if (i + 1 < nCh) {
      issue(i + 1, buf ^ 1);
      WAIT_ASYNC(4);        // chunk i complete (in-order), chunk i+1 in flight
    } else {
      WAIT_ASYNC(0);
    }
    __syncthreads();        // all waves' chunk-i tiles visible

    union FragU { v16bf v; uint4 q[2]; };
    FragU af[2], bfr[4];
    #pragma unroll
    for (int mf = 0; mf < 2; mf++) {
      const unsigned short* s = &la[buf][wm * 32 + mf * 16 + fl][kb];
      af[mf].q[0] = *(const uint4*)(s);
      af[mf].q[1] = *(const uint4*)(s + 16);
    }
    #pragma unroll
    for (int nf = 0; nf < 4; nf++) {
      const unsigned short* s = &lb[buf][wn * 64 + nf * 16 + fl][kb];
      bfr[nf].q[0] = *(const uint4*)(s);
      bfr[nf].q[1] = *(const uint4*)(s + 16);
    }

    #pragma unroll
    for (int mf = 0; mf < 2; mf++)
      #pragma unroll
      for (int nf = 0; nf < 4; nf++)
        acc[mf][nf] = __builtin_amdgcn_wmma_f32_16x16x32_bf16(
            false, af[mf].v, false, bfr[nf].v, (short)0, acc[mf][nf], false, false);

    __syncthreads();        // buf can be overwritten by iteration i+1's issue
  }

  // Epilogue: C layout = VGPR e -> row e + 8*(lane>=16), col = lane&15
  #pragma unroll
  for (int mf = 0; mf < 2; mf++) {
    #pragma unroll
    for (int nf = 0; nf < 4; nf++) {
      int gc = tileN + wn * 64 + nf * 16 + (lane & 15);
      #pragma unroll
      for (int e = 0; e < 8; e++) {
        int gr = tileM + wm * 32 + mf * 16 + e + ((lane >> 4) << 3);
        if (gr < Mrows) {
          float v = acc[mf][nf][e] + bias[gc];
          if (act) v = tanhf(v);
          long long idx = (long long)(gr / nn) * strideB
                        + (long long)(gr % nn) * strideRow + gc;
          if (outIsBf16) ((unsigned short*)Out)[idx] = f2bf(v);
          else           ((float*)Out)[idx] = v;
        }
      }
    }
  }
}

// ---------------------------------------------------------------------------
// Classifier: out[r, c] = tanh( [down|up] . clf_W[:,c] + clf_b[c] ), f32 out
// ---------------------------------------------------------------------------
__global__ __launch_bounds__(256)
void clf_kernel(const unsigned short* __restrict__ down,
                const unsigned short* __restrict__ up,
                const float* __restrict__ clfW, const float* __restrict__ clfb,
                float* __restrict__ out) {
  __shared__ float red[256][3];
  int r = blockIdx.x;                        // 0 .. B*N-1  ( == b*N + node )
  long long base = (long long)r * H_;
  float s0 = 0.f, s1 = 0.f, s2 = 0.f;
  for (int h = threadIdx.x; h < H_; h += 256) {
    float d = bf2f(down[base + h]);
    float u = bf2f(up[base + h]);
    const float* w0 = clfW + (long long)h * 3;
    const float* w1 = clfW + (long long)(H_ + h) * 3;
    s0 += d * w0[0] + u * w1[0];
    s1 += d * w0[1] + u * w1[1];
    s2 += d * w0[2] + u * w1[2];
  }
  red[threadIdx.x][0] = s0; red[threadIdx.x][1] = s1; red[threadIdx.x][2] = s2;
  __syncthreads();
  for (int off = 128; off > 0; off >>= 1) {
    if (threadIdx.x < off) {
      red[threadIdx.x][0] += red[threadIdx.x + off][0];
      red[threadIdx.x][1] += red[threadIdx.x + off][1];
      red[threadIdx.x][2] += red[threadIdx.x + off][2];
    }
    __syncthreads();
  }
  if (threadIdx.x < 3)
    out[(long long)r * 3 + threadIdx.x] = tanhf(red[0][threadIdx.x] + clfb[threadIdx.x]);
}

// ---------------------------------------------------------------------------
// Host orchestration
// ---------------------------------------------------------------------------
extern "C" void kernel_launch(void* const* d_in, const int* in_sizes, int n_in,
                              void* d_out, int out_size, void* d_ws, size_t ws_size,
                              hipStream_t stream) {
  (void)in_sizes; (void)n_in; (void)out_size; (void)ws_size;

  const int*   x     = (const int*)d_in[0];
  const int*   wi    = (const int*)d_in[1];
  const int*   cue   = (const int*)d_in[2];
  /* d_in[3] adj (unused), d_in[5] pre_embs (unused) */
  const int*   pos   = (const int*)d_in[4];
  const float* wordW = (const float*)d_in[6];
  const float* synW  = (const float*)d_in[7];
  const float* leafW = (const float*)d_in[8];
  const float* leafb = (const float*)d_in[9];
  const float* compW = (const float*)d_in[10];
  const float* compb = (const float*)d_in[11];
  const float* grevW = (const float*)d_in[12];
  const float* grevb = (const float*)d_in[13];
  const float* decW  = (const float*)d_in[14];
  const float* decb  = (const float*)d_in[15];
  const float* clfW  = (const float*)d_in[16];
  const float* clfb  = (const float*)d_in[17];

  // Workspace layout (bytes)
  char* ws = (char*)d_ws;
  const long long upBytes  = (long long)B_ * N_ * H_ * 2;     // 67,043,328
  const long long ctxBytes = 17825792LL * 2;                  // max(16384*1088, 8192*2176) halves
  const long long tmpBytes = 8192LL * 2048 * 2;               // dec output temp
  unsigned short* up     = (unsigned short*)(ws);
  unsigned short* down   = (unsigned short*)(ws + upBytes);
  unsigned short* ctx    = (unsigned short*)(ws + 2 * upBytes);
  unsigned short* tmp    = (unsigned short*)(ws + 2 * upBytes + ctxBytes);
  char* wt = ws + 2 * upBytes + ctxBytes + tmpBytes;
  unsigned short* leafWt = (unsigned short*)(wt);
  unsigned short* compWt = (unsigned short*)(wt + (long long)1024 * KL_P * 2);
  unsigned short* grevWt = (unsigned short*)(wt + (long long)1024 * KL_P * 2
                                                + (long long)1024 * KC_P * 2);
  unsigned short* decWt  = (unsigned short*)(wt + (long long)1024 * KL_P * 2
                                                + (long long)1024 * KC_P * 2
                                                + (long long)1024 * KG_P * 2);

  const long long sUpB = (long long)N_ * H_;   // batch stride inside up/down

  // 1) bf16 transposed weights
  {
    long long t;
    t = (long long)1024 * KL_P;
    wT_kernel<<<dim3((unsigned)((t + 255) / 256)), 256, 0, stream>>>(leafW, leafWt, 1077, 1024, KL_P);
    t = (long long)1024 * KC_P;
    wT_kernel<<<dim3((unsigned)((t + 255) / 256)), 256, 0, stream>>>(compW, compWt, 2098, 1024, KC_P);
    t = (long long)1024 * KG_P;
    wT_kernel<<<dim3((unsigned)((t + 255) / 256)), 256, 0, stream>>>(grevW, grevWt, 1074, 1024, KG_P);
    t = (long long)2048 * KD_P;
    wT_kernel<<<dim3((unsigned)((t + 255) / 256)), 256, 0, stream>>>(decW, decWt, 2148, 2048, KD_P);
  }

  // 2) Leaf embedding -> up[:, L-1 .. 2L-2]
  leaf_build<<<dim3(B_ * L_), 128, 0, stream>>>(x, wi, cue, pos, wordW, synW, ctx);
  gemm_bf16_wmma<<<dim3(1024 / TN, (B_ * L_ + TM - 1) / TM), 256, 0, stream>>>(
      ctx, leafWt, leafb, (void*)(up + (long long)(L_ - 1) * H_),
      B_ * L_, KL_P, /*act*/0, /*bf16*/1, L_, sUpB, H_);

  // 3) Upward pass
  for (int lvl = LVLS_ - 1; lvl >= 0; --lvl) {
    int nn = 1 << lvl, first = nn - 1, rows = B_ * nn;
    comp_build<<<dim3(rows), 128, 0, stream>>>(x, synW, up, ctx, first, nn);
    gemm_bf16_wmma<<<dim3(1024 / TN, (rows + TM - 1) / TM), 256, 0, stream>>>(
        ctx, compWt, compb, (void*)(up + (long long)first * H_),
        rows, KC_P, 1, 1, nn, sUpB, H_);
  }

  // 4) Root reversal -> down[:, 0]
  root_build<<<dim3(B_), 128, 0, stream>>>(x, synW, up, ctx);
  gemm_bf16_wmma<<<dim3(1024 / TN, 1), 256, 0, stream>>>(
      ctx, grevWt, grevb, (void*)down, B_, KG_P, 1, 1, 1, sUpB, 0);

  // 5) Downward pass
  for (int lvl = 0; lvl < LVLS_; ++lvl) {
    int nn = 1 << lvl, first = nn - 1, rows = B_ * nn;
    dec_build<<<dim3(rows), 128, 0, stream>>>(x, pos, synW, up, down, ctx,
                                              first, nn, lvl == LVLS_ - 1);
    gemm_bf16_wmma<<<dim3(2048 / TN, (rows + TM - 1) / TM), 256, 0, stream>>>(
        ctx, decWt, decb, (void*)tmp, rows, KD_P, 1, 1, 1, (long long)(2 * H_), 0);
    dec_scatter<<<dim3(rows), 256, 0, stream>>>(tmp, down, first, nn);
  }

  // 6) Classifier -> d_out (f32)
  clf_kernel<<<dim3(B_ * N_), 256, 0, stream>>>(down, up, clfW, clfb, (float*)d_out);
}
